// GCNModel_19524921327987
// MI455X (gfx1250) — matrix operand order repaired and checked by reading, not compiled, not run on hardware
//
#include <hip/hip_runtime.h>
#include <hip/hip_bf16.h>

typedef __attribute__((ext_vector_type(16))) _Float16 v16h;
typedef __attribute__((ext_vector_type(8)))  float    v8f;

#define NNODES  50000
#define NPAD    50016      // multiple of 16
#define NEDGES  800000
#define HID     128
#define NLAYERS 10

// ---------------- utility kernels ----------------

__global__ __launch_bounds__(256) void k_zero_i(int* __restrict__ p, int n) {
  int i = blockIdx.x * 256 + threadIdx.x;
  if (i < n) p[i] = 0;
}

__global__ __launch_bounds__(256) void k_deg(const int* __restrict__ dst,
                                             int* __restrict__ deg, int nE) {
  int e = blockIdx.x * 256 + threadIdx.x;
  if (e < nE) atomicAdd(&deg[dst[e]], 1);
}

__global__ __launch_bounds__(256) void k_invdeg(const int* __restrict__ deg,
                                                float* __restrict__ inv, int n) {
  int i = blockIdx.x * 256 + threadIdx.x;
  if (i < n) { int d = deg[i]; inv[i] = (d > 0) ? (1.f / (float)d) : 0.f; }
}

// ---------------- exclusive scan (degree -> CSR offsets), 3 kernels ----------------

__global__ __launch_bounds__(256) void k_scan1(const int* __restrict__ deg,
                                               int* __restrict__ offs,
                                               int* __restrict__ bsum, int n) {
  __shared__ int tmp[256];
  int gid = blockIdx.x * 256 + threadIdx.x;
  int v = (gid < n) ? deg[gid] : 0;
  tmp[threadIdx.x] = v;
  __syncthreads();
  for (int off = 1; off < 256; off <<= 1) {
    int t = (threadIdx.x >= off) ? tmp[threadIdx.x - off] : 0;
    __syncthreads();
    tmp[threadIdx.x] += t;
    __syncthreads();
  }
  if (gid < n) offs[gid] = tmp[threadIdx.x] - v;        // exclusive
  if (threadIdx.x == 255) bsum[blockIdx.x] = tmp[255];  // block total
}

__global__ __launch_bounds__(256) void k_scan2(int* __restrict__ bsum, int nb) {
  __shared__ int tmp[256];
  int v = (threadIdx.x < nb) ? bsum[threadIdx.x] : 0;
  tmp[threadIdx.x] = v;
  __syncthreads();
  for (int off = 1; off < 256; off <<= 1) {
    int t = (threadIdx.x >= off) ? tmp[threadIdx.x - off] : 0;
    __syncthreads();
    tmp[threadIdx.x] += t;
    __syncthreads();
  }
  if (threadIdx.x < nb) bsum[threadIdx.x] = tmp[threadIdx.x] - v;  // exclusive
}

__global__ __launch_bounds__(256) void k_scan3(int* __restrict__ offs,
                                               const int* __restrict__ bsum, int n) {
  int gid = blockIdx.x * 256 + threadIdx.x;
  if (gid < n) offs[gid] += bsum[blockIdx.x];
  if (gid == 0) offs[NPAD] = NEDGES;   // total in-degree = all edges
}

__global__ __launch_bounds__(256) void k_fill(const int* __restrict__ src,
                                              const int* __restrict__ dst,
                                              const int* __restrict__ offs,
                                              int* __restrict__ cursor,
                                              int* __restrict__ srclist, int nE) {
  int e = blockIdx.x * 256 + threadIdx.x;
  if (e >= nE) return;
  int d = dst[e];
  int p = atomicAdd(&cursor[d], 1);
  srclist[offs[d] + p] = src[e];
}

// Pack W_gcn (row-major [L][K=128][N=128] f32) into the per-lane WMMA B-operand
// image, f16:  wp[((l*4+kb)*8+n)*32*16 + lane*16 + j]
__global__ __launch_bounds__(256) void k_pack_w(const float* __restrict__ W,
                                                _Float16* __restrict__ wp, int total) {
  int idx = blockIdx.x * 256 + threadIdx.x;
  if (idx >= total) return;
  int j    =  idx        & 15;
  int lane = (idx >> 4)  & 31;
  int n    = (idx >> 9)  & 7;
  int kb   = (idx >> 12) & 3;
  int l    =  idx >> 14;
  int k   = kb * 32 + ((lane & 16) >> 1) + (j & 7) + ((j & 8) << 1);
  int col = n * 16 + (lane & 15);
  wp[idx] = (_Float16)W[(size_t)l * HID * HID + (size_t)k * HID + col];
}

// ---------------- encoder: h = relu(node_feats @ W_enc + b_enc) ----------------

__global__ __launch_bounds__(256) void k_encoder(const float* __restrict__ nf,
                                                 const float* __restrict__ We,
                                                 const float* __restrict__ be,
                                                 float* __restrict__ h) {
  int idx = blockIdx.x * 256 + threadIdx.x;
  if (idx >= NPAD * HID) return;
  int node = idx >> 7, c = idx & 127;
  float acc = 0.f;
  if (node < NNODES) {
    acc = be[c];
    const float* f = nf + (size_t)node * 9;
#pragma unroll
    for (int i = 0; i < 9; ++i) acc = fmaf(f[i], We[i * HID + c], acc);
    acc = fmaxf(acc, 0.f);
  }
  h[idx] = acc;   // pad rows forced to zero
}

// ---------------- WMMA GEMM: m = h @ W + b  (one wave = 16x128 tile) ----------------

__global__ __launch_bounds__(256) void k_gemm_wmma(const float* __restrict__ h,
                                                   const _Float16* __restrict__ wp,
                                                   const float* __restrict__ bias,
                                                   float* __restrict__ m) {
  int wave = (blockIdx.x * 256 + (int)threadIdx.x) >> 5;
  int lane = threadIdx.x & 31;
  if (wave >= NPAD / 16) return;          // wave-uniform exit: EXEC all-ones for WMMA

  const float* hrow = h + (size_t)(wave * 16 + (lane & 15)) * HID;
  const v16h*  wl   = (const v16h*)wp;

  v8f acc[8] = {};
#pragma unroll
  for (int kb = 0; kb < 4; ++kb) {
    // A layout (16-bit, 16x32): lanes 0-15 hold K{0..7,16..23}, lanes 16-31 K{8..15,24..31}
    int k0 = kb * 32 + ((lane & 16) >> 1);          // +8 for the upper half-wave
    float4 x0 = *(const float4*)(hrow + k0);
    float4 x1 = *(const float4*)(hrow + k0 + 4);
    float4 x2 = *(const float4*)(hrow + k0 + 16);
    float4 x3 = *(const float4*)(hrow + k0 + 20);
    v16h a;
    a[0]  = (_Float16)x0.x; a[1]  = (_Float16)x0.y; a[2]  = (_Float16)x0.z; a[3]  = (_Float16)x0.w;
    a[4]  = (_Float16)x1.x; a[5]  = (_Float16)x1.y; a[6]  = (_Float16)x1.z; a[7]  = (_Float16)x1.w;
    a[8]  = (_Float16)x2.x; a[9]  = (_Float16)x2.y; a[10] = (_Float16)x2.z; a[11] = (_Float16)x2.w;
    a[12] = (_Float16)x3.x; a[13] = (_Float16)x3.y; a[14] = (_Float16)x3.z; a[15] = (_Float16)x3.w;
#pragma unroll
    for (int n = 0; n < 8; ++n) {
      v16h b = wl[(kb * 8 + n) * 32 + lane];        // contiguous 32B per lane
      acc[n] = __builtin_amdgcn_wmma_f32_16x16x32_f16(
          /*neg_a=*/false, a, /*neg_b=*/false, b,
          /*c_mod=*/(short)0, acc[n], /*reuse_a=*/false, /*reuse_b=*/false);
    }
  }
  // C/D layout: VGPR v -> row (lane<16 ? v : 8+v), col = n*16 + lane%16
  int rbase = wave * 16 + ((lane & 16) >> 1);
  int colb  = lane & 15;
#pragma unroll
  for (int n = 0; n < 8; ++n) {
    int col = n * 16 + colb;
    float bv = bias[col];
#pragma unroll
    for (int v = 0; v < 8; ++v)
      m[(size_t)(rbase + v) * HID + col] = acc[n][v] + bv;
  }
}

// ---------------- pull aggregation: h[n] = relu(inv_deg[n] * sum_{e->n} m[src]) ----------------
// one wave32 per node; lane = 4-channel chunk -> each edge visit reads a
// fully-coalesced 512B row of m (L2-resident). No atomics, no zero pass.

__global__ __launch_bounds__(256) void k_gather(const float* __restrict__ m,
                                                const int* __restrict__ offs,
                                                const int* __restrict__ srclist,
                                                const float* __restrict__ inv,
                                                float* __restrict__ h) {
  int wave = (blockIdx.x * 256 + (int)threadIdx.x) >> 5;
  int lane = threadIdx.x & 31;
  if (wave >= NPAD) return;
  int node = wave;
  int c = lane << 2;
  int beg = offs[node], end = offs[node + 1];
  float4 acc = make_float4(0.f, 0.f, 0.f, 0.f);
  for (int i = beg; i < end; ++i) {
    int s = srclist[i];
    if (i + 1 < end)
      __builtin_prefetch((const void*)(m + (size_t)srclist[i + 1] * HID + c), 0, 0);
    float4 v = *(const float4*)(m + (size_t)s * HID + c);
    acc.x += v.x; acc.y += v.y; acc.z += v.z; acc.w += v.w;
  }
  float sc = inv[node];   // 0 for zero-degree and pad nodes
  float4 o;
  o.x = fmaxf(acc.x * sc, 0.f);
  o.y = fmaxf(acc.y * sc, 0.f);
  o.z = fmaxf(acc.z * sc, 0.f);
  o.w = fmaxf(acc.w * sc, 0.f);
  *(float4*)(h + (size_t)node * HID + c) = o;
}

// ---------------- decoder: logits + softmax per edge ----------------

__global__ __launch_bounds__(256) void k_decoder(const float* __restrict__ h,
                                                 const float* __restrict__ ef,
                                                 const int* __restrict__ src,
                                                 const int* __restrict__ dst,
                                                 const float* __restrict__ Wd,
                                                 const float* __restrict__ bd,
                                                 float* __restrict__ logits,
                                                 float* __restrict__ prob) {
  __shared__ float sW[270 * 2];
  __shared__ float sb[2];
  for (int i = threadIdx.x; i < 540; i += 256) sW[i] = Wd[i];
  if (threadIdx.x < 2) sb[threadIdx.x] = bd[threadIdx.x];
  __syncthreads();

  int e = blockIdx.x * 256 + threadIdx.x;
  if (e >= NEDGES) return;
  const float* hs = h + (size_t)src[e] * HID;
  const float* hd = h + (size_t)dst[e] * HID;
  const float* ep = ef + (size_t)e * 14;
  float l0 = sb[0], l1 = sb[1];
#pragma unroll 4
  for (int i = 0; i < HID; ++i) { float v = hs[i]; l0 = fmaf(v, sW[2*i],           l0); l1 = fmaf(v, sW[2*i+1],           l1); }
#pragma unroll 4
  for (int i = 0; i < HID; ++i) { float v = hd[i]; l0 = fmaf(v, sW[2*(HID+i)],     l0); l1 = fmaf(v, sW[2*(HID+i)+1],     l1); }
#pragma unroll
  for (int i = 0; i < 14; ++i)  { float v = ep[i]; l0 = fmaf(v, sW[2*(2*HID+i)],   l0); l1 = fmaf(v, sW[2*(2*HID+i)+1],   l1); }
  logits[(size_t)e * 2]     = l0;
  logits[(size_t)e * 2 + 1] = l1;
  float mx = fmaxf(l0, l1);
  float e0 = __expf(l0 - mx), e1 = __expf(l1 - mx);
  float r  = 1.f / (e0 + e1);
  prob[(size_t)e * 2]     = e0 * r;
  prob[(size_t)e * 2 + 1] = e1 * r;
}

// ---------------- host launch ----------------

extern "C" void kernel_launch(void* const* d_in, const int* in_sizes, int n_in,
                              void* d_out, int out_size, void* d_ws, size_t ws_size,
                              hipStream_t stream) {
  const float* node_feats = (const float*)d_in[0];
  const float* edge_feats = (const float*)d_in[1];
  const int*   src        = (const int*)  d_in[2];
  const int*   dst        = (const int*)  d_in[3];
  const float* W_enc      = (const float*)d_in[4];
  const float* b_enc      = (const float*)d_in[5];
  const float* W_gcn      = (const float*)d_in[6];
  const float* b_gcn      = (const float*)d_in[7];
  const float* W_dec      = (const float*)d_in[8];
  const float* b_dec      = (const float*)d_in[9];

  float* logits = (float*)d_out;
  float* prob   = logits + (size_t)NEDGES * 2;

  char* p = (char*)d_ws;
  auto take = [&](size_t bytes) -> char* {
    char* r = p; p += (bytes + 255) & ~(size_t)255; return r;
  };
  int*      deg     = (int*)     take((size_t)NPAD * 4);
  int*      offs    = (int*)     take((size_t)(NPAD + 1) * 4);
  int*      bsum    = (int*)     take((size_t)256 * 4);
  int*      cursor  = (int*)     take((size_t)NPAD * 4);
  int*      srclist = (int*)     take((size_t)NEDGES * 4);
  float*    inv     = (float*)   take((size_t)NPAD * 4);
  float*    h       = (float*)   take((size_t)NPAD * HID * 4);
  float*    m       = (float*)   take((size_t)NPAD * HID * 4);
  _Float16* wp      = (_Float16*)take((size_t)NLAYERS * HID * HID * 2);

  const int T  = 256;
  const int NH = NPAD * HID;
  const int nb = (NPAD + T - 1) / T;   // 196 scan blocks (<= 256)

  // degree, inverse degree
  k_zero_i<<<nb, T, 0, stream>>>(deg, NPAD);
  k_deg   <<<(NEDGES + T - 1) / T, T, 0, stream>>>(dst, deg, NEDGES);
  k_invdeg<<<nb, T, 0, stream>>>(deg, inv, NPAD);

  // CSR by dst: exclusive scan + atomic-cursor fill
  k_scan1<<<nb, T, 0, stream>>>(deg, offs, bsum, NPAD);
  k_scan2<<<1,  T, 0, stream>>>(bsum, nb);
  k_scan3<<<nb, T, 0, stream>>>(offs, bsum, NPAD);
  k_zero_i<<<nb, T, 0, stream>>>(cursor, NPAD);
  k_fill  <<<(NEDGES + T - 1) / T, T, 0, stream>>>(src, dst, offs, cursor, srclist, NEDGES);

  // pack GCN weights into WMMA B-operand layout (f16)
  int wtotal = NLAYERS * HID * HID;
  k_pack_w<<<(wtotal + T - 1) / T, T, 0, stream>>>(W_gcn, wp, wtotal);

  // encoder
  k_encoder<<<(NH + T - 1) / T, T, 0, stream>>>(node_feats, W_enc, b_enc, h);

  // 10 GCN layers: WMMA GEMM then pull-aggregation (fused relu * inv_deg)
  int nwaves      = NPAD / 16;                      // 3126 row-tiles
  int gemm_blocks = (nwaves * 32 + T - 1) / T;
  int gath_blocks = (NPAD * 32 + T - 1) / T;        // one wave per node
  for (int l = 0; l < NLAYERS; ++l) {
    k_gemm_wmma<<<gemm_blocks, T, 0, stream>>>(h, wp + (size_t)l * HID * HID,
                                               b_gcn + (size_t)l * HID, m);
    k_gather<<<gath_blocks, T, 0, stream>>>(m, offs, srclist, inv, h);
  }

  // edge decoder + softmax
  k_decoder<<<(NEDGES + T - 1) / T, T, 0, stream>>>(h, edge_feats, src, dst,
                                                    W_dec, b_dec, logits, prob);
}
